// SparseNerfDepthLoss_36163624632854
// MI455X (gfx1250) — matrix operand order repaired and checked by reading, not compiled, not run on hardware
//
#include <hip/hip_runtime.h>
#include <stdint.h>

// Problem constants (match reference)
#define HH 1024
#define WW 2048
#define PS 16
#define STR 8
#define NH ((HH - PS) / STR + 1)   // 127
#define NW ((WW - PS) / STR + 1)   // 255
#define P_TOT (NH * NW)            // 32385 patches
#define N_EL (PS * PS)             // 256 elements per patch
#define MARGIN 1.0e-4f
#define NBUCKET 256                // atomic reduction buckets in d_ws

typedef __attribute__((ext_vector_type(16))) _Float16 v16h;
typedef __attribute__((ext_vector_type(8)))  float    v8f;

// ---- CDNA5 helpers -------------------------------------------------------

// LDS byte offset of a __shared__ object: generic (flat) LDS addresses are
// {aperture_hi, lds_offset}; truncation to 32 bits yields the DS offset.
__device__ __forceinline__ uint32_t lds_off32(const void* p) {
    return (uint32_t)(uintptr_t)p;
}

// Per-lane asynchronous gather, global -> LDS (gfx1250 ASYNCcnt path, GV mode).
__device__ __forceinline__ void async_gather_b32(uint32_t lds_byte_off,
                                                 const void* gaddr) {
    asm volatile("global_load_async_to_lds_b32 %0, %1, off"
                 :
                 : "v"(lds_byte_off), "v"((uint64_t)(uintptr_t)gaddr)
                 : "memory");
}

__device__ __forceinline__ void wait_asynccnt0() {
    asm volatile("s_wait_asynccnt 0" ::: "memory");
}

__device__ __forceinline__ unsigned long long shflxor_u64(unsigned long long v,
                                                          int lanemask) {
    int lo = (int)(uint32_t)v;
    int hi = (int)(uint32_t)(v >> 32);
    lo = __shfl_xor(lo, lanemask, 32);
    hi = __shfl_xor(hi, lanemask, 32);
    return ((unsigned long long)(uint32_t)hi << 32) | (uint32_t)lo;
}

// ---- kernels -------------------------------------------------------------

__global__ void init_ws_kernel(float* ws) {
    int t = blockIdx.x * blockDim.x + threadIdx.x;
    if (t < 2 * NBUCKET) ws[t] = 0.0f;
}

__global__ __launch_bounds__(N_EL)
void patch_loss_kernel(const float* __restrict__ pred,
                       const float* __restrict__ targ,
                       const unsigned char* __restrict__ mask,  // jax bool = 1B
                       const float* __restrict__ noise,
                       float* __restrict__ ws_sum,   // [NBUCKET]
                       float* __restrict__ ws_cnt)   // [NBUCKET]
{
    __shared__ float              s_pred[N_EL];
    __shared__ float              s_targ[N_EL];
    __shared__ int                s_scan[N_EL];     // inclusive cumsum(mask)
    __shared__ unsigned long long s_key [N_EL];
    __shared__ uint32_t           s_bits[8];        // 256-bit mask bitmap
    __shared__ float              s_red [16];       // per-wave partials

    const int t  = threadIdx.x;            // element index within patch
    const int p  = blockIdx.x;             // patch index
    const int ph = p / NW;
    const int pw = p - ph * NW;
    const int pr = t >> 4;
    const int pc = t & 15;
    const int pix = (ph * STR + pr) * WW + (pw * STR + pc);
    const int w   = t >> 5;                // wave id (0..7)

    // Async DMA of the patch's pred/target into LDS; overlaps the ballot,
    // the WMMA scan and the whole bitonic sort below.
    async_gather_b32(lds_off32(&s_pred[t]), pred + pix);
    async_gather_b32(lds_off32(&s_targ[t]), targ + pix);

    const int   m  = mask[pix] ? 1 : 0;
    const float nz = noise[(size_t)p * N_EL + t];

    // ---- publish mask as a 256-bit bitmap (8 ballot words)
    uint32_t bal = (uint32_t)__ballot(m != 0);
    if ((t & 31) == 0) s_bits[w] = bal;
    __syncthreads();

    // ---- rank = cumsum(mask)-1 via WMMA scan-as-matmul (wave 0 only).
    // D(16x16,f32) = A(16x32,f16 lower-triangular ones, K>=16 zero)
    //              x B(32x16,f16; B[k][n] = maskbit(n*16+k), K>=16 zero)
    // => D[m][n] = inclusive scan inside segment n; add exclusive prefix of
    // column sums (D[15][n]) for the cross-segment offset. Counts <=16 are
    // exact in f16; offsets <=256 exact in f32.
    if (t < 32) {
        const int L = t;
        const int g = L >> 4;         // lane group
        const int mrow = L & 15;

        // B fragment: lanes 0-15 hold column n=L, K slot s maps to K=s.
        uint32_t bits16 = (s_bits[L >> 1] >> ((L & 1) * 16)) & 0xFFFFu;
        v16h a, b;
        #pragma unroll
        for (int s = 0; s < 16; ++s) {
            // A: lane L holds M=mrow; K = g*16+s
            int k = g * 16 + s;
            a[s] = (k <= mrow) ? (_Float16)1.0f : (_Float16)0.0f;
            b[s] = (g == 0 && ((bits16 >> s) & 1u)) ? (_Float16)1.0f
                                                    : (_Float16)0.0f;
        }
        v8f cz = {};
        v8f d = __builtin_amdgcn_wmma_f32_16x16x32_f16(
            false, a, false, b, (short)0, cz, false, false);

        // exclusive prefix of the 16 column sums (D[15][n] @ lane 16+n, VGPR7)
        const int n = L & 15;
        float d7 = d[7];
        float off = 0.0f;
        #pragma unroll
        for (int i = 0; i < 15; ++i) {
            float cs = __shfl(d7, 16 + i, 32);
            if (i < n) off += cs;
        }
        // D layout: lane L -> n = L&15, m = g*8 + r (VGPR r)
        #pragma unroll
        for (int r = 0; r < 8; ++r)
            s_scan[n * 16 + g * 8 + r] = (int)(d[r] + off);
    }
    __syncthreads();

    const int rank         = s_scan[t] - 1;
    const int total_masked = s_scan[N_EL - 1];

    // ---- stable argsort(noise | +inf): register bitonic on (bits<<32)|idx.
    // j<32 stages exchange via __shfl_xor (wave-local, no barrier);
    // only the 6 stages with j>=32 round-trip through LDS.
    const uint32_t kb = m ? __float_as_uint(nz) : 0x7F800000u;  // +inf unmasked
    unsigned long long key = ((unsigned long long)kb << 32) | (unsigned)t;

    for (unsigned k = 2; k <= N_EL; k <<= 1) {
        for (unsigned j = k >> 1; j > 0; j >>= 1) {
            unsigned long long pk;
            if (j >= 32) {
                s_key[t] = key;
                __syncthreads();
                pk = s_key[t ^ j];
                __syncthreads();
            } else {
                pk = shflxor_u64(key, (int)j);
            }
            const bool asc   = ((t & k) == 0);
            const bool lower = ((t & j) == 0);
            key = (asc == lower) ? (key < pk ? key : pk)
                                 : (key > pk ? key : pk);
        }
    }

    // Publish sorted keys; same barrier also orders each wave's drained
    // async copies (ASYNCcnt is per-wave) before anyone reads s_pred/s_targ.
    s_key[t] = key;
    wait_asynccnt0();
    __syncthreads();

    // ---- partner pairing + sign-mismatch loss term
    const int rk      = (rank < 0) ? 0 : rank;                       // clip
    const int partner = (int)(unsigned)(s_key[rk] & 0xFFFFFFFFull);

    const float dtv = s_targ[t] - s_targ[partner];
    const float dpv = s_pred[t] - s_pred[partner] + MARGIN;
    const int sdt = (dtv > 0.0f) - (dtv < 0.0f);                     // jnp.sign
    const int sdp = (dpv > 0.0f) - (dpv < 0.0f);
    const bool sel = (m != 0) && (sdt != sdp);
    float v1 = sel ? fabsf(dpv) : 0.0f;   // sum |dp|
    float v2 = sel ? 1.0f : 0.0f;         // count

    // ---- reduction: 5 shfl_xor steps per wave, then 8-entry LDS combine
    #pragma unroll
    for (int o = 16; o >= 1; o >>= 1) {
        v1 += __shfl_xor(v1, o, 32);
        v2 += __shfl_xor(v2, o, 32);
    }
    if ((t & 31) == 0) { s_red[2 * w] = v1; s_red[2 * w + 1] = v2; }
    __syncthreads();

    if (t == 0 && total_masked > 0) {
        float sum = 0.0f, cnt = 0.0f;
        #pragma unroll
        for (int i = 0; i < 8; ++i) { sum += s_red[2 * i]; cnt += s_red[2 * i + 1]; }
        float loss = sum / cnt;  // 0/0 -> NaN exactly as reference
        atomicAdd(&ws_sum[p & (NBUCKET - 1)], loss);
        atomicAdd(&ws_cnt[p & (NBUCKET - 1)], 1.0f);
    }
}

__global__ __launch_bounds__(NBUCKET)
void finalize_kernel(const float* __restrict__ ws_sum,
                     const float* __restrict__ ws_cnt,
                     float* __restrict__ out) {
    __shared__ float a[NBUCKET];
    __shared__ float b[NBUCKET];
    int t = threadIdx.x;
    a[t] = ws_sum[t];
    b[t] = ws_cnt[t];
    __syncthreads();
    for (int s = NBUCKET / 2; s > 0; s >>= 1) {
        if (t < s) { a[t] += a[t + s]; b[t] += b[t + s]; }
        __syncthreads();
    }
    if (t == 0) out[0] = a[0] / b[0];
}

// ---- host entry ----------------------------------------------------------

extern "C" void kernel_launch(void* const* d_in, const int* in_sizes, int n_in,
                              void* d_out, int out_size, void* d_ws, size_t ws_size,
                              hipStream_t stream) {
    const float*         pred  = (const float*)d_in[0];
    const float*         targ  = (const float*)d_in[1];
    const unsigned char* mask  = (const unsigned char*)d_in[2];  // bool, 1B/elem
    const float*         noise = (const float*)d_in[3];

    float* ws     = (float*)d_ws;            // needs 2*NBUCKET*4 = 2 KB
    float* ws_sum = ws;
    float* ws_cnt = ws + NBUCKET;

    hipLaunchKernelGGL(init_ws_kernel, dim3(1), dim3(2 * NBUCKET), 0, stream, ws);
    hipLaunchKernelGGL(patch_loss_kernel, dim3(P_TOT), dim3(N_EL), 0, stream,
                       pred, targ, mask, noise, ws_sum, ws_cnt);
    hipLaunchKernelGGL(finalize_kernel, dim3(1), dim3(NBUCKET), 0, stream,
                       ws_sum, ws_cnt, (float*)d_out);
}
